// URanker_1726576857797
// MI455X (gfx1250) — matrix operand order repaired and checked by reading, not compiled, not run on hardware
//
#include <hip/hip_runtime.h>
#include <hip/hip_bf16.h>
#include <math.h>

typedef __attribute__((ext_vector_type(16))) __bf16 v16bf;
typedef __attribute__((ext_vector_type(8)))  __bf16 v8bf;
typedef __attribute__((ext_vector_type(8)))  float  v8f;
typedef __attribute__((ext_vector_type(4)))  float  v4f;

constexpr int kB   = 8;
constexpr int kN   = 3137;
constexpr int kC   = 768;
constexpr int kH   = 8;       // heads
constexpr int kCh  = 96;      // per-head channels
constexpr int kM   = kB * kN; // 25096 rows
constexpr int kQC  = 3 * kC;  // 2304
constexpr int kHW  = 56;
constexpr int kP   = kHW * kHW; // 3136
constexpr float kScale = 0.10206207261596575f; // 96^-0.5

// ---------------------------------------------------------------------------
// WMMA fragment gathers from LDS (bf16 16x16x32, wave32 layouts per ISA 7.12.2)
// A 16x32: lane<16 -> M=lane, K in {0..7, 16..23}; lane>=16 -> M=lane-16, K in {8..15, 24..31}
// B 32x16 (stored transposed [n][k] in LDS): lane<16 -> N=lane, K in {0..7, 16..23}; lane>=16 -> K in {8..15, 24..31}
// ---------------------------------------------------------------------------
__device__ inline v16bf frag_a(const __bf16* tile, int stride, int lane) {
  int m  = lane & 15;
  int kg = lane >> 4;
  const __bf16* row = tile + m * stride;
  v8bf lo = *(const v8bf*)(row + kg * 8);
  v8bf hi = *(const v8bf*)(row + 16 + kg * 8);
  return __builtin_shufflevector(lo, hi, 0,1,2,3,4,5,6,7,8,9,10,11,12,13,14,15);
}
__device__ inline v16bf frag_b(const __bf16* tileT, int stride, int lane) {
  int n  = lane & 15;
  int kh = lane >> 4;
  const __bf16* p = tileT + n * stride + kh * 8;
  v8bf lo = *(const v8bf*)(p);
  v8bf hi = *(const v8bf*)(p + 16);
  return __builtin_shufflevector(lo, hi, 0,1,2,3,4,5,6,7,8,9,10,11,12,13,14,15);
}

// ---------------------------------------------------------------------------
// bf16-WMMA GEMM: Out[M,Nout] = A[M,K] * Bw[K,Nout] (+ bias), f32 I/O.
// Block 256 = 8 waves; tile 128x128; waves in 4x2 grid, each owns 32x64
// (2 row tiles x 4 col tiles -> 8 wmma per K-chunk, 12 ds_b128 per chunk).
// Software pipeline: LDS double-buffer; per chunk issue next chunk's global
// loads first, run wmma on current LDS buffer (DS-only waits; loads in
// flight), then convert+store to the other buffer; one barrier per chunk.
// Branchless staging: row index clamped (garbage rows only affect their own
// store-guarded output rows).
// ---------------------------------------------------------------------------
constexpr int kBM = 128, kBN = 128, kBK = 32, kAST = 40, kBST = 40;

__device__ inline void gemm_load_global(const float* __restrict__ A,
                                        const float* __restrict__ Bw,
                                        int M, int K, int Nout,
                                        int m0, int n0, int k0, int tid,
                                        v4f* va, v4f* vb) {
#pragma unroll
  for (int i = 0; i < 4; ++i) {
    int idx = tid + i * 256;            // 1024 float4 slots (128x32)
    int r   = idx >> 3;
    int c4  = (idx & 7) * 4;
    int gr  = min(m0 + r, M - 1);
    va[i] = *(const v4f*)(A + (size_t)gr * K + k0 + c4);
  }
#pragma unroll
  for (int i = 0; i < 4; ++i) {
    int idx = tid + i * 256;            // 1024 float4 slots (32x128)
    int r   = idx >> 5;
    int c4  = (idx & 31) * 4;
    vb[i] = *(const v4f*)(Bw + (size_t)(k0 + r) * Nout + n0 + c4);
  }
}

__device__ inline void gemm_store_lds(const v4f* va, const v4f* vb, int tid,
                                      __bf16* __restrict__ As,
                                      __bf16* __restrict__ Bt) {
#pragma unroll
  for (int i = 0; i < 4; ++i) {
    int idx = tid + i * 256;
    int r   = idx >> 3;
    int c4  = (idx & 7) * 4;
    __bf16* dst = As + r * kAST + c4;
    dst[0] = (__bf16)va[i][0]; dst[1] = (__bf16)va[i][1];
    dst[2] = (__bf16)va[i][2]; dst[3] = (__bf16)va[i][3];
  }
#pragma unroll
  for (int i = 0; i < 4; ++i) {
    int idx = tid + i * 256;
    int r   = idx >> 5;
    int c4  = (idx & 31) * 4;
    Bt[(c4 + 0) * kBST + r] = (__bf16)vb[i][0];
    Bt[(c4 + 1) * kBST + r] = (__bf16)vb[i][1];
    Bt[(c4 + 2) * kBST + r] = (__bf16)vb[i][2];
    Bt[(c4 + 3) * kBST + r] = (__bf16)vb[i][3];
  }
}

__global__ __launch_bounds__(256) void gemm_bf16_wmma(
    const float* __restrict__ A, const float* __restrict__ Bw,
    const float* __restrict__ bias, float* __restrict__ Out,
    int M, int K, int Nout)
{
  __shared__ alignas(16) __bf16 As[2][kBM * kAST];
  __shared__ alignas(16) __bf16 Bt[2][kBN * kBST];
  const int tid  = threadIdx.x;
  const int lane = tid & 31;
  const int wave = tid >> 5;
  const int rw = wave >> 1;   // 0..3 : 32-row group
  const int cw = wave & 1;    // 0..1 : 64-col group
  const int m0 = blockIdx.y * kBM;
  const int n0 = blockIdx.x * kBN;

  v8f acc[2][4] = {};
  const int nk = K / kBK;

  // Prologue: stage chunk 0 into buffer 0
  {
    v4f va[4], vb[4];
    gemm_load_global(A, Bw, M, K, Nout, m0, n0, 0, tid, va, vb);
    gemm_store_lds(va, vb, tid, As[0], Bt[0]);
  }

  for (int kc = 0; kc < nk; ++kc) {
    __syncthreads();                    // buf[cur] ready for everyone
    const int cur = kc & 1;
    v4f va[4], vb[4];
    const bool more = (kc + 1) < nk;
    if (more)                           // issue next chunk's loads early
      gemm_load_global(A, Bw, M, K, Nout, m0, n0, (kc + 1) * kBK, tid, va, vb);

    // Compute on buf[cur]: DS waits only; global loads stay in flight
    v16bf af0 = frag_a(As[cur] + (rw * 32     ) * kAST, kAST, lane);
    v16bf af1 = frag_a(As[cur] + (rw * 32 + 16) * kAST, kAST, lane);
#pragma unroll
    for (int nt = 0; nt < 4; ++nt) {
      v16bf bf = frag_b(Bt[cur] + (cw * 64 + nt * 16) * kBST, kBST, lane);
      acc[0][nt] = __builtin_amdgcn_wmma_f32_16x16x32_bf16(
          false, af0, false, bf, (short)0, acc[0][nt], false, false);
      acc[1][nt] = __builtin_amdgcn_wmma_f32_16x16x32_bf16(
          false, af1, false, bf, (short)0, acc[1][nt], false, false);
    }

    if (more)                           // now wait loads, convert, fill buf[cur^1]
      gemm_store_lds(va, vb, tid, As[cur ^ 1], Bt[cur ^ 1]);
  }

  const int nlane = lane & 15;
  for (int mt = 0; mt < 2; ++mt) {
    int mbase = m0 + rw * 32 + mt * 16 + (lane >> 4) * 8;
    for (int nt = 0; nt < 4; ++nt) {
      int gc = n0 + cw * 64 + nt * 16 + nlane;
      float bv = bias ? bias[gc] : 0.0f;
      for (int r = 0; r < 8; ++r) {
        int gr = mbase + r;
        if (gr < M) Out[(size_t)gr * Nout + gc] = acc[mt][nt][r] + bv;
      }
    }
  }
}

// ---------------------------------------------------------------------------
// Column softmax of k over N (per b,h,ch) -- 3-stage (partial, reduce, write)
// ---------------------------------------------------------------------------
constexpr int kSplit = 32;
constexpr int kChunkTok = (kN + kSplit - 1) / kSplit; // 99

__global__ void ksm_partial(const float* __restrict__ qkv,
                            float* __restrict__ pm, float* __restrict__ ps)
{
  int bh = blockIdx.x, sp = blockIdx.y, ch = threadIdx.x;
  if (ch < kCh) {
    int b = bh >> 3, h = bh & 7;
    const float* base = qkv + (size_t)b * kN * kQC + kC + h * kCh + ch;
    int lo = sp * kChunkTok;
    int hi = min(lo + kChunkTok, kN);
    float m = -__builtin_inff(), s = 0.0f;
    for (int n = lo; n < hi; ++n) {
      float x = base[(size_t)n * kQC];
      float mn = fmaxf(m, x);
      s = s * __expf(m - mn) + __expf(x - mn);
      m = mn;
    }
    pm[(bh * kSplit + sp) * kCh + ch] = m;
    ps[(bh * kSplit + sp) * kCh + ch] = s;
  }
}

__global__ void ksm_reduce(const float* __restrict__ pm, const float* __restrict__ ps,
                           float* __restrict__ wm, float* __restrict__ winv)
{
  int bh = blockIdx.x, ch = threadIdx.x;
  if (ch < kCh) {
    float m = -__builtin_inff(), s = 0.0f;
    for (int sp = 0; sp < kSplit; ++sp) {
      float pmi = pm[(bh * kSplit + sp) * kCh + ch];
      float psi = ps[(bh * kSplit + sp) * kCh + ch];
      float mn = fmaxf(m, pmi);
      s = s * __expf(m - mn) + psi * __expf(pmi - mn);
      m = mn;
    }
    wm[bh * kCh + ch] = m;
    winv[bh * kCh + ch] = 1.0f / s;
  }
}

__global__ void ksm_write(const float* __restrict__ qkv,
                          const float* __restrict__ wm, const float* __restrict__ winv,
                          float* __restrict__ ksm)
{
  size_t t = (size_t)blockIdx.x * blockDim.x + threadIdx.x;
  size_t total = (size_t)kB * kH * kN * kCh;
  if (t >= total) return;
  int ch = (int)(t % kCh);
  size_t r = t / kCh;
  int n  = (int)(r % kN);
  int bh = (int)(r / kN);
  int b = bh >> 3, h = bh & 7;
  float x = qkv[((size_t)b * kN + n) * kQC + kC + h * kCh + ch];
  ksm[t] = __expf(x - wm[bh * kCh + ch]) * winv[bh * kCh + ch];
}

// ---------------------------------------------------------------------------
// kv = k_softmax^T @ v : 64 batched 96x96 GEMMs, K = N = 3137 (chunks of 32).
// OOB K-tokens must contribute zero -> clamped load + branchless select.
// ---------------------------------------------------------------------------
__global__ __launch_bounds__(256) void kv_wmma(const float* __restrict__ ksm,
                                               const float* __restrict__ qkv,
                                               float* __restrict__ kvbuf)
{
  constexpr int ST = 40;
  __shared__ alignas(16) __bf16 Kt[kCh * ST]; // [k-channel][token]
  __shared__ alignas(16) __bf16 Vt[kCh * ST]; // [v-channel][token]
  const int tid = threadIdx.x, lane = tid & 31, wave = tid >> 5;
  const int bh = blockIdx.x;
  const int b = bh >> 3, h = bh & 7;

  int tiles[5]; int ntile = 0;
  for (int t = wave; t < 36; t += 8) tiles[ntile++] = t;
  v8f acc[5] = {};

  for (int c0 = 0; c0 < kN; c0 += 32) {
    for (int i = tid; i < 32 * kCh; i += 256) {
      int tok = i / kCh, ch = i % kCh;
      bool ok = (c0 + tok) < kN;
      int tt = min(c0 + tok, kN - 1);
      float kvval = ksm[((size_t)bh * kN + tt) * kCh + ch];
      float vvval = qkv[((size_t)b * kN + tt) * kQC + 2 * kC + h * kCh + ch];
      kvval = ok ? kvval : 0.0f;
      vvval = ok ? vvval : 0.0f;
      Kt[ch * ST + tok] = (__bf16)kvval;
      Vt[ch * ST + tok] = (__bf16)vvval;
    }
    __syncthreads();
    for (int i = 0; i < ntile; ++i) {
      int kt = tiles[i] / 6, vt = tiles[i] % 6;
      v16bf af = frag_a(Kt + kt * 16 * ST, ST, lane);
      v16bf bf = frag_b(Vt + vt * 16 * ST, ST, lane);
      acc[i] = __builtin_amdgcn_wmma_f32_16x16x32_bf16(
          false, af, false, bf, (short)0, acc[i], false, false);
    }
    __syncthreads();
  }

  for (int i = 0; i < ntile; ++i) {
    int kt = tiles[i] / 6, vt = tiles[i] % 6;
    int gm = kt * 16 + (lane >> 4) * 8;
    int gn = vt * 16 + (lane & 15);
    for (int r = 0; r < 8; ++r)
      kvbuf[((size_t)bh * kCh + gm + r) * kCh + gn] = acc[i][r];
  }
}

// ---------------------------------------------------------------------------
// factor_att = SCALE * (q @ kv) : per (b,h), M=3137 K=96 Nout=96.
// OOB tokens only pollute their own (store-guarded) rows -> clamp, no branch.
// ---------------------------------------------------------------------------
__global__ __launch_bounds__(256) void fatt_wmma(const float* __restrict__ qkv,
                                                 const float* __restrict__ kvbuf,
                                                 float* __restrict__ fa)
{
  constexpr int ST = 104;
  __shared__ alignas(16) __bf16 Qs [128 * ST];  // [token][k-ch]
  __shared__ alignas(16) __bf16 KVt[kCh * ST];  // [v-ch][k-ch]
  const int tid = threadIdx.x, lane = tid & 31, wave = tid >> 5;
  const int bh = blockIdx.y;
  const int b = bh >> 3, h = bh & 7;
  const int t0 = blockIdx.x * 128;

  for (int i = tid; i < 128 * kCh; i += 256) {
    int tok = i / kCh, ch = i % kCh;
    int tt = min(t0 + tok, kN - 1);
    float q = qkv[((size_t)b * kN + tt) * kQC + h * kCh + ch];
    Qs[tok * ST + ch] = (__bf16)q;
  }
  for (int i = tid; i < kCh * kCh; i += 256) {
    int kc = i / kCh, vc = i % kCh;
    KVt[vc * ST + kc] = (__bf16)kvbuf[((size_t)bh * kCh + kc) * kCh + vc];
  }
  __syncthreads();

  v8f acc[6] = {};
  for (int kc = 0; kc < 3; ++kc) {
    v16bf af = frag_a(Qs + wave * 16 * ST + kc * 32, ST, lane);
    for (int vt = 0; vt < 6; ++vt) {
      v16bf bf = frag_b(KVt + vt * 16 * ST + kc * 32, ST, lane);
      acc[vt] = __builtin_amdgcn_wmma_f32_16x16x32_bf16(
          false, af, false, bf, (short)0, acc[vt], false, false);
    }
  }

  const int mbase = t0 + wave * 16 + (lane >> 4) * 8;
  const int nlane = lane & 15;
  for (int vt = 0; vt < 6; ++vt) {
    int gc = h * kCh + vt * 16 + nlane;
    for (int r = 0; r < 8; ++r) {
      int tok = mbase + r;
      if (tok < kN)
        fa[((size_t)b * kN + tok) * kC + gc] = kScale * acc[vt][r];
    }
  }
}

// ---------------------------------------------------------------------------
// Extract v image tokens into pixel-major NCHW layout (coalesced conv input)
// ---------------------------------------------------------------------------
__global__ void extract_v(const float* __restrict__ qkv, float* __restrict__ vimg)
{
  size_t t = (size_t)blockIdx.x * blockDim.x + threadIdx.x;
  size_t total = (size_t)kB * kC * kP;
  if (t >= total) return;
  int c = (int)(t % kC);
  size_t r = t / kC;
  int p = (int)(r % kP);
  int b = (int)(r / kP);
  vimg[((size_t)b * kC + c) * kP + p] =
      qkv[((size_t)b * kN + 1 + p) * kQC + 2 * kC + c];
}

// ---------------------------------------------------------------------------
// Depthwise conv, window 3/5/7 by channel band; one channel per block.x
// ---------------------------------------------------------------------------
__global__ __launch_bounds__(256) void dwconv_kernel(
    const float* __restrict__ vimg,
    const float* __restrict__ w3, const float* __restrict__ b3,
    const float* __restrict__ w5, const float* __restrict__ b5,
    const float* __restrict__ w7, const float* __restrict__ b7,
    float* __restrict__ convv)
{
  int bc = blockIdx.x;
  int c = bc % kC;
  int p = blockIdx.y * 256 + threadIdx.x;
  if (p >= kP) return;
  int y = p / kHW, x = p % kHW;
  const float* img = vimg + (size_t)bc * kP;

  int ksz; const float* w; float bias;
  if (c < 192)      { ksz = 3; w = w3 + c * 9;          bias = b3[c]; }
  else if (c < 480) { ksz = 5; w = w5 + (c - 192) * 25; bias = b5[c - 192]; }
  else              { ksz = 7; w = w7 + (c - 480) * 49; bias = b7[c - 480]; }
  int pad = ksz >> 1;

  float acc = bias;
  for (int ky = 0; ky < ksz; ++ky) {
    int yy = y + ky - pad;
    if (yy < 0 || yy >= kHW) continue;
    for (int kx = 0; kx < ksz; ++kx) {
      int xx = x + kx - pad;
      if (xx < 0 || xx >= kHW) continue;
      acc += w[ky * ksz + kx] * img[yy * kHW + xx];
    }
  }
  convv[(size_t)bc * kP + p] = acc;
}

// ---------------------------------------------------------------------------
// CRPE combine (in place on fa): fa[b,n,c] += q[b,n,c] * conv_v  (n >= 1)
// ---------------------------------------------------------------------------
__global__ void combine_crpe(float* __restrict__ fa,
                             const float* __restrict__ qkv,
                             const float* __restrict__ convv)
{
  size_t t = (size_t)blockIdx.x * blockDim.x + threadIdx.x;
  size_t total = (size_t)kM * kC;
  if (t >= total) return;
  int c = (int)(t % kC);
  size_t r = t / kC;
  int n = (int)(r % kN);
  int b = (int)(r / kN);
  float val = fa[t];
  if (n >= 1) {
    int p = n - 1;
    float q = qkv[((size_t)b * kN + n) * kQC + c];
    val += q * convv[((size_t)b * kC + c) * kP + p];
  }
  fa[t] = val;
}

// ---------------------------------------------------------------------------
extern "C" void kernel_launch(void* const* d_in, const int* in_sizes, int n_in,
                              void* d_out, int out_size, void* d_ws, size_t ws_size,
                              hipStream_t stream) {
  (void)in_sizes; (void)n_in; (void)out_size; (void)ws_size;
  const float* x     = (const float*)d_in[0];
  const float* Wqkv  = (const float*)d_in[1];
  const float* Wproj = (const float*)d_in[2];
  const float* bproj = (const float*)d_in[3];
  const float* cw3   = (const float*)d_in[4];
  const float* cb3   = (const float*)d_in[5];
  const float* cw5   = (const float*)d_in[6];
  const float* cb5   = (const float*)d_in[7];
  const float* cw7   = (const float*)d_in[8];
  const float* cb7   = (const float*)d_in[9];
  float* out = (float*)d_out;

  // workspace carve-up (floats)
  float* w = (float*)d_ws;
  const size_t SZ_QKV  = (size_t)kM * kQC;            // 57.8M
  const size_t SZ_KSM  = (size_t)kB * kH * kN * kCh;  // 19.3M
  const size_t SZ_PART = (size_t)kB * kH * kSplit * kCh;
  const size_t SZ_STAT = (size_t)kB * kH * kCh;
  const size_t SZ_KV   = (size_t)kB * kH * kCh * kCh;
  const size_t SZ_FA   = (size_t)kM * kC;             // 19.3M
  const size_t SZ_IMG  = (size_t)kB * kC * kP;        // 19.3M

  float* qkv   = w;                  w += SZ_QKV;
  float* ksm   = w;                  w += SZ_KSM;
  float* pm    = w;                  w += SZ_PART;
  float* ps    = w;                  w += SZ_PART;
  float* wmax  = w;                  w += SZ_STAT;
  float* winv  = w;                  w += SZ_STAT;
  float* kvbuf = w;                  w += SZ_KV;
  float* fa    = w;                  w += SZ_FA;
  float* vimg  = w;                  w += SZ_IMG;
  float* convv = w;                  w += SZ_IMG;

  const int mblocks = (kM + 127) / 128; // 197

  // 1) qkv = x @ Wqkv                (bf16 WMMA, 128x128 tiles, pipelined)
  gemm_bf16_wmma<<<dim3(kQC / 128, mblocks), 256, 0, stream>>>(
      x, Wqkv, nullptr, qkv, kM, kC, kQC);

  // 2) column softmax of k over N
  ksm_partial<<<dim3(kB * kH, kSplit), 128, 0, stream>>>(qkv, pm, ps);
  ksm_reduce<<<kB * kH, 128, 0, stream>>>(pm, ps, wmax, winv);
  {
    size_t total = SZ_KSM;
    ksm_write<<<(unsigned)((total + 255) / 256), 256, 0, stream>>>(qkv, wmax, winv, ksm);
  }

  // 3) kv = k_softmax^T @ v          (bf16 WMMA, batched)
  kv_wmma<<<kB * kH, 256, 0, stream>>>(ksm, qkv, kvbuf);

  // 4) fa = SCALE * q @ kv           (bf16 WMMA, batched) -> (b,n,c) layout
  fatt_wmma<<<dim3((kN + 127) / 128, kB * kH), 256, 0, stream>>>(qkv, kvbuf, fa);

  // 5) depthwise convs on v image
  {
    size_t total = SZ_IMG;
    extract_v<<<(unsigned)((total + 255) / 256), 256, 0, stream>>>(qkv, vimg);
  }
  dwconv_kernel<<<dim3(kB * kC, (kP + 255) / 256), 256, 0, stream>>>(
      vimg, cw3, cb3, cw5, cb5, cw7, cb7, convv);

  // 6) fa += q * conv_v (CRPE), in place
  {
    size_t total = SZ_FA;
    combine_crpe<<<(unsigned)((total + 255) / 256), 256, 0, stream>>>(fa, qkv, convv);
  }

  // 7) out = fa @ Wproj + bproj      (bf16 WMMA, 128x128 tiles, pipelined)
  gemm_bf16_wmma<<<dim3(kC / 128, mblocks), 256, 0, stream>>>(
      fa, Wproj, bproj, out, kM, kC, kC);
}
